// DlrmTop_6116033429805
// MI455X (gfx1250) — compile-verified, matrix operand order checked by reading, not run on hardware
//
#include <hip/hip_runtime.h>
#include <hip/hip_bf16.h>

typedef __attribute__((ext_vector_type(16))) __bf16 v16bf;
typedef __attribute__((ext_vector_type(8)))  __bf16 v8bf;
typedef __attribute__((ext_vector_type(8)))  float  v8f;
typedef __attribute__((ext_vector_type(4)))  unsigned int v4u;
typedef __attribute__((ext_vector_type(8)))  int v8i;
typedef __attribute__((ext_vector_type(4)))  int v4i;

static constexpr int    BATCH = 32768;
static constexpr int    NI    = 27;
static constexpr int    EMB   = 128;
static constexpr int    K0P   = 512;   // 480 padded to 512

// workspace layout (bytes)
static constexpr size_t OFF_XBF = 0;                                     // B*512 bf16  (X, later Y3)
static constexpr size_t OFF_A   = OFF_XBF + (size_t)BATCH * 512 * 2;     // B*1024 bf16 (Y1, later Y4)
static constexpr size_t OFF_B   = OFF_A   + (size_t)BATCH * 1024 * 2;    // B*1024 bf16 (Y2)
static constexpr size_t OFF_W0  = OFF_B   + (size_t)BATCH * 1024 * 2;    // 1024*512 bf16
static constexpr size_t OFF_W1  = OFF_W0  + (size_t)1024 * 512 * 2;      // 1024*1024 bf16
static constexpr size_t OFF_W2  = OFF_W1  + (size_t)1024 * 1024 * 2;     // 512*1024 bf16
static constexpr size_t OFF_W3  = OFF_W2  + (size_t)512 * 1024 * 2;      // 256*512 bf16
static constexpr size_t OFF_W4  = OFF_W3  + (size_t)256 * 512 * 2;       // 256 bf16

// ---------------- weight packing ----------------
__global__ __launch_bounds__(256) void pack_bf16(const float* __restrict__ src,
                                                 __bf16* __restrict__ dst, int n) {
    int i = blockIdx.x * 256 + threadIdx.x;
    if (i < n) dst[i] = (__bf16)src[i];
}

__global__ __launch_bounds__(256) void pack_w0(const float* __restrict__ src,
                                               __bf16* __restrict__ dst) {
    int i = blockIdx.x * 256 + threadIdx.x;      // over 1024*512
    int n = i >> 9, k = i & 511;
    dst[i] = (k < 480) ? (__bf16)src[n * 480 + k] : (__bf16)0.0f;
}

// ---------------- interaction: X = [bottom[:,0,:], tril(A A^T), 0, pad] -------------
__device__ __forceinline__ v16bf make_frag_f32(const float* __restrict__ A, int m, int k0) {
    v16bf f;
    if (m < NI) {
        const float* p = A + m * EMB + k0;
        float4 x0 = *(const float4*)(p);
        float4 x1 = *(const float4*)(p + 4);
        float4 y0 = *(const float4*)(p + 16);
        float4 y1 = *(const float4*)(p + 20);
        f[0]=(__bf16)x0.x; f[1]=(__bf16)x0.y; f[2]=(__bf16)x0.z; f[3]=(__bf16)x0.w;
        f[4]=(__bf16)x1.x; f[5]=(__bf16)x1.y; f[6]=(__bf16)x1.z; f[7]=(__bf16)x1.w;
        f[8]=(__bf16)y0.x; f[9]=(__bf16)y0.y; f[10]=(__bf16)y0.z; f[11]=(__bf16)y0.w;
        f[12]=(__bf16)y1.x; f[13]=(__bf16)y1.y; f[14]=(__bf16)y1.z; f[15]=(__bf16)y1.w;
    } else {
        #pragma unroll
        for (int j = 0; j < 16; ++j) f[j] = (__bf16)0.0f;
    }
    return f;
}

__global__ __launch_bounds__(256) void interact_kernel(const float* __restrict__ bot,
                                                       __bf16* __restrict__ Xb) {
    const int wave = threadIdx.x >> 5;
    const int lane = threadIdx.x & 31;
    const int b    = blockIdx.x * 8 + wave;
    const int hf   = lane >> 4;        // half-wave
    const int lm   = lane & 15;

    const float* A = bot + (size_t)b * NI * EMB;

    v8f c00 = {}, c10 = {}, c11 = {};
    #pragma unroll
    for (int kb = 0; kb < EMB; kb += 32) {
        const int k0 = kb + hf * 8;
        v16bf f0 = make_frag_f32(A, lm, k0);        // rows 0..15
        v16bf f1 = make_frag_f32(A, 16 + lm, k0);   // rows 16..31 (>=27 zero)
        // G = A * A^T : B-operand fragment == A-operand fragment of that row tile
        c00 = __builtin_amdgcn_wmma_f32_16x16x32_bf16(false, f0, false, f0, (short)0, c00, false, false);
        c10 = __builtin_amdgcn_wmma_f32_16x16x32_bf16(false, f1, false, f0, (short)0, c10, false, false);
        c11 = __builtin_amdgcn_wmma_f32_16x16x32_bf16(false, f1, false, f1, (short)0, c11, false, false);
    }

    __bf16* Xrow = Xb + (size_t)b * K0P;

    // bottom_mlp_output = A[0][:]: 128 floats, 4 per lane
    {
        float4 v = *(const float4*)(A + lane * 4);
        Xrow[lane * 4 + 0] = (__bf16)v.x;
        Xrow[lane * 4 + 1] = (__bf16)v.y;
        Xrow[lane * 4 + 2] = (__bf16)v.z;
        Xrow[lane * 4 + 3] = (__bf16)v.w;
    }
    // zero pad: element 479 + K-padding 480..511
    for (int i = 479 + lane; i < K0P; i += 32) Xrow[i] = (__bf16)0.0f;

    // scatter strict lower triangle: idx(r,c) = r*(r-1)/2 + c
    #pragma unroll
    for (int i = 0; i < 8; ++i) {
        {   // C00: r in 0..15, c in 0..15
            int r = hf * 8 + i, c = lm;
            if (c < r) Xrow[EMB + r * (r - 1) / 2 + c] = (__bf16)c00[i];
        }
        {   // C10: r in 16..31, c in 0..15
            int r = 16 + hf * 8 + i, c = lm;
            if (r < NI) Xrow[EMB + r * (r - 1) / 2 + c] = (__bf16)c10[i];
        }
        {   // C11: r in 16..31, c in 16..31
            int r = 16 + hf * 8 + i, c = 16 + lm;
            if (r < NI && c < r) Xrow[EMB + r * (r - 1) / 2 + c] = (__bf16)c11[i];
        }
    }
}

// ---------------- bf16 WMMA GEMM: Y = act(X * W^T + bias) ----------------
// X: [B x K] bf16 row-major, W: [N x K] bf16 row-major, Y: [B x N] bf16.
// Block: 128 batch rows x 64 cols; 8 waves, each wave 16 rows x 64 cols.
// Weight tile (64 rows x 32 K, bf16) staged into LDS by the Tensor Data Mover:
//   one tensor_load_to_lds per K-step, issued by wave 0, tracked on TENSORcnt.
// D# padding: every 16 DWORDs (one 32-bf16 row) insert 8 DWORDs -> LDS row
// stride 48 bf16 = 96 B (16B-aligned fragment reads, spread banks).
template <int K, int N, bool RELU>
__global__ __launch_bounds__(256) void gemm_bf16(const __bf16* __restrict__ X,
                                                 const __bf16* __restrict__ W,
                                                 const float*  __restrict__ bias,
                                                 __bf16* __restrict__ Y) {
    __shared__ __align__(16) __bf16 ldsW[64][48];

    const int tid  = threadIdx.x;
    const int wave = tid >> 5;
    const int lane = tid & 31;
    const int hf   = lane >> 4;
    const int lm   = lane & 15;

    const int rowBase = blockIdx.x * 128 + wave * 16;
    const int colBase = blockIdx.y * 64;

    v8f acc[4];
    #pragma unroll
    for (int t = 0; t < 4; ++t) acc[t] = (v8f){};

    const unsigned int lds_base = (unsigned int)(uintptr_t)(&ldsW[0][0]);
    const unsigned long long wbase = (unsigned long long)(uintptr_t)(W + (size_t)colBase * K);
    const __bf16* xrow = X + (size_t)(rowBase + lm) * K + hf * 8;

    for (int kb = 0; kb < K; kb += 32) {
        __syncthreads();   // previous tile fully consumed
        if (wave == 0) {
            const unsigned long long ga = wbase + (unsigned long long)kb * 2ull;
            // D# group 0: count=1 | lds_addr | global_addr | type=2
            v4u g0 = { 1u,
                       lds_base,
                       (unsigned int)ga,
                       (unsigned int)((ga >> 32) & 0x1FFFFFFu) | 0x80000000u };
            // D# group 1: data_size=2B, pad every 16 DW by 8 DW,
            // tensor_dim0=K, tensor_dim1=64, tile=32x64, dim0_stride=K
            v8i g1 = { (int)((1u << 16) | (1u << 20) | (3u << 22) | (7u << 25)),
                       (int)(((unsigned int)K & 0xffffu) << 16),
                       (int)(((unsigned int)K >> 16) | (64u << 16)),
                       (int)(32u << 16),
                       (int)64,
                       (int)K,
                       0, 0 };
            v4i gz4 = { 0, 0, 0, 0 };
            v8i gz8 = { 0, 0, 0, 0, 0, 0, 0, 0 };
            __builtin_amdgcn_tensor_load_to_lds(g0, g1, gz4, gz4, gz8, 0);
            __builtin_amdgcn_s_wait_tensorcnt(0);
        }
        __syncthreads();   // tile visible to all waves

        if (kb + 64 < K) __builtin_prefetch(xrow + kb + 64, 0, 0);

        v8bf alo = *(const v8bf*)(xrow + kb);
        v8bf ahi = *(const v8bf*)(xrow + kb + 16);
        v16bf a = __builtin_shufflevector(alo, ahi, 0,1,2,3,4,5,6,7,8,9,10,11,12,13,14,15);

        #pragma unroll
        for (int t = 0; t < 4; ++t) {
            const __bf16* wr = &ldsW[t * 16 + lm][hf * 8];
            v8bf blo = *(const v8bf*)(wr);
            v8bf bhi = *(const v8bf*)(wr + 16);
            v16bf bb = __builtin_shufflevector(blo, bhi, 0,1,2,3,4,5,6,7,8,9,10,11,12,13,14,15);
            acc[t] = __builtin_amdgcn_wmma_f32_16x16x32_bf16(false, a, false, bb, (short)0, acc[t], false, false);
        }
    }

    #pragma unroll
    for (int t = 0; t < 4; ++t) {
        const int col = colBase + t * 16 + lm;
        const float bv = bias[col];
        #pragma unroll
        for (int i = 0; i < 8; ++i) {
            const int row = rowBase + hf * 8 + i;
            float v = acc[t][i] + bv;
            if (RELU) v = v > 0.0f ? v : 0.0f;
            Y[(size_t)row * N + col] = (__bf16)v;
        }
    }
}

// ---------------- final 256 -> 1 layer (fp32) ----------------
__global__ __launch_bounds__(256) void final_dot(const __bf16* __restrict__ Y,
                                                 const __bf16* __restrict__ w,
                                                 const float*  __restrict__ b4,
                                                 float* __restrict__ out) {
    const int i = blockIdx.x * 256 + threadIdx.x;
    const __bf16* row = Y + (size_t)i * 256;
    float acc = 0.0f;
    #pragma unroll 4
    for (int k = 0; k < 256; k += 8) {
        v8bf rv = *(const v8bf*)(row + k);
        v8bf wv = *(const v8bf*)(w + k);
        #pragma unroll
        for (int j = 0; j < 8; ++j) acc += (float)rv[j] * (float)wv[j];
    }
    out[i] = acc + b4[0];
}

extern "C" void kernel_launch(void* const* d_in, const int* in_sizes, int n_in,
                              void* d_out, int out_size, void* d_ws, size_t ws_size,
                              hipStream_t stream) {
    const float* bot = (const float*)d_in[0];
    const float* W0  = (const float*)d_in[1];  const float* b0 = (const float*)d_in[2];
    const float* W1  = (const float*)d_in[3];  const float* b1 = (const float*)d_in[4];
    const float* W2  = (const float*)d_in[5];  const float* b2 = (const float*)d_in[6];
    const float* W3  = (const float*)d_in[7];  const float* b3 = (const float*)d_in[8];
    const float* W4  = (const float*)d_in[9];  const float* b4 = (const float*)d_in[10];
    float* out = (float*)d_out;

    char* ws = (char*)d_ws;
    __bf16* Xbf = (__bf16*)(ws + OFF_XBF);
    __bf16* bufA = (__bf16*)(ws + OFF_A);
    __bf16* bufB = (__bf16*)(ws + OFF_B);
    __bf16* W0p = (__bf16*)(ws + OFF_W0);
    __bf16* W1p = (__bf16*)(ws + OFF_W1);
    __bf16* W2p = (__bf16*)(ws + OFF_W2);
    __bf16* W3p = (__bf16*)(ws + OFF_W3);
    __bf16* W4p = (__bf16*)(ws + OFF_W4);

    // pack weights to bf16 (W0 zero-padded K 480->512)
    pack_w0<<<(1024 * 512) / 256, 256, 0, stream>>>(W0, W0p);
    pack_bf16<<<(1024 * 1024) / 256, 256, 0, stream>>>(W1, W1p, 1024 * 1024);
    pack_bf16<<<(512 * 1024) / 256, 256, 0, stream>>>(W2, W2p, 512 * 1024);
    pack_bf16<<<(256 * 512) / 256, 256, 0, stream>>>(W3, W3p, 256 * 512);
    pack_bf16<<<1, 256, 0, stream>>>(W4, W4p, 256);

    // feature construction (interaction via WMMA)
    interact_kernel<<<BATCH / 8, 256, 0, stream>>>(bot, Xbf);

    // MLP layers
    gemm_bf16<512, 1024, true><<<dim3(BATCH / 128, 1024 / 64), 256, 0, stream>>>(Xbf, W0p, b0, bufA);
    gemm_bf16<1024, 1024, true><<<dim3(BATCH / 128, 1024 / 64), 256, 0, stream>>>(bufA, W1p, b1, bufB);
    gemm_bf16<1024, 512, true><<<dim3(BATCH / 128, 512 / 64), 256, 0, stream>>>(bufB, W2p, b2, Xbf);
    gemm_bf16<512, 256, true><<<dim3(BATCH / 128, 256 / 64), 256, 0, stream>>>(Xbf, W3p, b3, bufA);

    // final 256 -> 1
    final_dot<<<BATCH / 256, 256, 0, stream>>>(bufA, W4p, b4, out);
}
